// ANI_50319836840320
// MI455X (gfx1250) — matrix operand order repaired and checked by reading, not compiled, not run on hardware
//
#include <hip/hip_runtime.h>
#include <hip/hip_bf16.h>
#include <stdint.h>

typedef __attribute__((ext_vector_type(16))) _Float16 v16h;
typedef __attribute__((ext_vector_type(8)))  _Float16 v8h;
typedef __attribute__((ext_vector_type(4)))  _Float16 v4h;
typedef __attribute__((ext_vector_type(8)))  float    v8f;

#define B_   32
#define N_   512
#define D_   384
#define E_   4
#define H1_  256
#define H2_  256
#define BN_  (B_*N_)
#define TM_  128          // atom rows per workgroup tile

// LDS strides in halves; all row strides are multiples of 8 halves (16B) so
// every WMMA fragment access is a 16B-aligned ds_load_b128 / async b128.
#define SX_  40           // Xs  [128][SX_]
#define SW_  40           // Ws  [256][SW_]  (transposed weight chunk: [col][k])
#define SH_  264          // h1s [128][SH_]

#define XS_OFF   (TM_*SH_)                   // halves
#define WS_OFF   (XS_OFF + TM_*SX_)          // halves (two Ws buffers follow)
#define ROWE_B   ((WS_OFF + 2*H1_*SW_)*2)    // bytes
#define SMEM_BYTES (ROWE_B + TM_*4 + 16)

#define WT1_ELEMS (E_*H1_*D_)    // 393216 f16
#define WT2_ELEMS (E_*H2_*H1_)   // 262144 f16

__device__ __forceinline__ float ssp(float x) {
  // shifted softplus: log(1+e^x) - log(2), numerically stable
  float ax = fabsf(x);
  return fmaxf(x, 0.0f) + log1pf(__expf(-ax)) - 0.69314718055994530942f;
}

__device__ __forceinline__ v16h cat8(v8h lo, v8h hi) {
  return __builtin_shufflevector(lo, hi, 0,1,2,3,4,5,6,7,8,9,10,11,12,13,14,15);
}

// Async LDS-DMA: copy one 16B chunk per lane of a 256x32-half weight tile.
// Wt is f16, pre-transposed: row n holds all K for output column n.
__device__ __forceinline__ void issue_w_async(const _Float16* __restrict__ Wt,
                                              int rowBase, int Kdim, int k0,
                                              _Float16* wsbuf, int tid) {
  #pragma unroll
  for (int it = 0; it < 4; ++it) {
    int slot = tid + it*256;          // 1024 x 16B chunks = 256 rows x 64B
    int n = slot >> 2, ch = slot & 3;
    unsigned long long g =
        (unsigned long long)(uintptr_t)(Wt + (size_t)(rowBase + n)*Kdim + k0 + ch*8);
    uint32_t lds = (uint32_t)(uintptr_t)(wsbuf + n*SW_ + ch*8);
    asm volatile("global_load_async_to_lds_b128 %0, %1, off"
                 :: "v"(lds), "v"(g) : "memory");
  }
}
__device__ __forceinline__ void wait_async_le4() {
  asm volatile("s_wait_asynccnt 0x4" ::: "memory");
}
__device__ __forceinline__ void wait_async_0() {
  asm volatile("s_wait_asynccnt 0x0" ::: "memory");
}

__global__ void zero_f32(float* __restrict__ p, int n) {
  int i = blockIdx.x * blockDim.x + threadIdx.x;
  if (i < n) p[i] = 0.0f;
}

// One-time: f32 weights -> f16, transposed to [e][out_col][k] for async tiles.
__global__ void prep_wt(const float* __restrict__ W1, const float* __restrict__ W2,
                        _Float16* __restrict__ W1t, _Float16* __restrict__ W2t) {
  int i = blockIdx.x * blockDim.x + threadIdx.x;
  if (i < WT1_ELEMS) {
    int n = i % H1_;
    int k = (i / H1_) % D_;
    int e = i / (H1_ * D_);
    W1t[((size_t)(e*H1_ + n))*D_ + k] = (_Float16)W1[i];
  }
  int j = i - WT1_ELEMS;
  if (j >= 0 && j < WT2_ELEMS) {
    int n = j % H2_;
    int k = (j / H2_) % H1_;
    int e = j / (H2_ * H1_);
    W2t[((size_t)(e*H2_ + n))*H1_ + k] = (_Float16)W2[j];
  }
}

template <bool WT>
__global__ __launch_bounds__(256, 1)
void ani_mlp(const float* __restrict__ Xg,     // [BN][D]
             const int*   __restrict__ Zg,     // [BN] atomic numbers
             const int*   __restrict__ Eg,     // [E]  element table
             const float* __restrict__ W1g, const float* __restrict__ b1g,
             const float* __restrict__ W2g, const float* __restrict__ b2g,
             const float* __restrict__ W3g, const float* __restrict__ b3g,
             const _Float16* __restrict__ W1t, const _Float16* __restrict__ W2t,
             float* __restrict__ outg)         // [B]
{
  extern __shared__ char smem[];
  _Float16* h1s  = (_Float16*)smem;             // [TM_][SH_] f16 hidden
  _Float16* Xs   = (_Float16*)smem + XS_OFF;    // [TM_][SX_] f16 input chunk
  _Float16* Ws0  = (_Float16*)smem + WS_OFF;    // [256][SW_] double-buffered
  _Float16* Ws1  = Ws0 + H1_*SW_;
  float*    row_e = (float*)(smem + ROWE_B);    // [TM_] per-atom dot partials
  __shared__ float bsum;

  const int tid   = threadIdx.x;
  const int lane  = tid & 31;
  const int wid   = tid >> 5;     // 0..7
  const int wrow  = wid >> 2;     // 0..1  -> 64 rows each
  const int wcol  = wid & 3;      // 0..3  -> 64 cols each
  const int mlane = lane & 15;
  const int hhi   = lane >> 4;    // half-wave select per CDNA5 WMMA layout

  const int e       = blockIdx.y;
  const int tileRow = blockIdx.x * TM_;

  if (tid < TM_) row_e[tid] = 0.0f;
  if (tid == 0)  bsum = 0.0f;

  if (WT) issue_w_async(W1t, e*H1_, D_, 0, Ws0, tid);   // prefetch tile 0

  v8f zeroacc = {};
  v8f c[4][4];
  #pragma unroll
  for (int i = 0; i < 4; ++i)
    #pragma unroll
    for (int j = 0; j < 4; ++j) c[i][j] = zeroacc;

  // ---------------- layer 1: [128 x 384] x [384 x 256] ----------------
  for (int kk = 0; kk < D_/32; ++kk) {
    const int k0 = kk * 32;
    _Float16* wsbuf = (WT && (kk & 1)) ? Ws1 : Ws0;
    __syncthreads();                       // all readers of prev buffers done
    if (WT) {
      if (kk+1 < D_/32)
        issue_w_async(W1t, e*H1_, D_, k0 + 32, (kk & 1) ? Ws0 : Ws1, tid);
    } else {
      // fallback: stage W1 chunk transposed from f32
      #pragma unroll
      for (int it = 0; it < 8; ++it) {
        int idx = (tid + it*256) * 4;      // over 32x256, n fastest
        int n = idx & 255, k = idx >> 8;
        float4 f = *(const float4*)(W1g + ((size_t)e*D_ + k0 + k)*H1_ + n);
        Ws0[(n+0)*SW_ + k] = (_Float16)f.x;
        Ws0[(n+1)*SW_ + k] = (_Float16)f.y;
        Ws0[(n+2)*SW_ + k] = (_Float16)f.z;
        Ws0[(n+3)*SW_ + k] = (_Float16)f.w;
      }
    }
    // stage X chunk 128x32 f32->f16 (1024 float4, 4 per thread)
    #pragma unroll
    for (int it = 0; it < 4; ++it) {
      int idx = tid + it*256;
      int r = idx >> 3, c4 = idx & 7;
      float4 f = *(const float4*)(Xg + (size_t)(tileRow + r)*D_ + k0 + c4*4);
      v4h h; h[0]=(_Float16)f.x; h[1]=(_Float16)f.y; h[2]=(_Float16)f.z; h[3]=(_Float16)f.w;
      *(v4h*)(Xs + r*SX_ + c4*4) = h;
    }
    if (kk+1 < D_/32 && tid < TM_)
      __builtin_prefetch(Xg + (size_t)(tileRow + tid)*D_ + k0 + 32, 0, 1);
    if (WT) { if (kk+1 < D_/32) wait_async_le4(); else wait_async_0(); }
    __syncthreads();

    v16h a[4];
    #pragma unroll
    for (int i = 0; i < 4; ++i) {
      const _Float16* p = Xs + (wrow*64 + i*16 + mlane)*SX_ + hhi*8;
      a[i] = cat8(*(const v8h*)p, *(const v8h*)(p + 16));
    }
    #pragma unroll
    for (int j = 0; j < 4; ++j) {
      const _Float16* q = wsbuf + (wcol*64 + j*16 + mlane)*SW_ + hhi*8;
      v16h b = cat8(*(const v8h*)q, *(const v8h*)(q + 16));
      #pragma unroll
      for (int i = 0; i < 4; ++i)
        c[i][j] = __builtin_amdgcn_wmma_f32_16x16x32_f16(
            false, a[i], false, b, (short)0, c[i][j], false, false);
    }
  }

  // prefetch first W2 tile while activations run (Ws0 free: last read kk=10)
  if (WT) issue_w_async(W2t, e*H2_, H1_, 0, Ws0, tid);

  // bias + shifted softplus, park h1 in LDS as f16
  #pragma unroll
  for (int j = 0; j < 4; ++j) {
    int n = wcol*64 + j*16 + mlane;
    float bj = b1g[e*H1_ + n];
    #pragma unroll
    for (int i = 0; i < 4; ++i) {
      int rb = wrow*64 + i*16 + hhi*8;
      #pragma unroll
      for (int v = 0; v < 8; ++v)
        h1s[(rb + v)*SH_ + n] = (_Float16)ssp(c[i][j][v] + bj);
    }
  }

  #pragma unroll
  for (int i = 0; i < 4; ++i)
    #pragma unroll
    for (int j = 0; j < 4; ++j) c[i][j] = zeroacc;

  // ---------------- layer 2: [128 x 256] x [256 x 256] ----------------
  for (int kk = 0; kk < H1_/32; ++kk) {
    const int k0 = kk * 32;
    _Float16* wsbuf = (WT && (kk & 1)) ? Ws1 : Ws0;
    __syncthreads();                       // also fences h1s writes (kk==0)
    if (WT) {
      if (kk+1 < H1_/32)
        issue_w_async(W2t, e*H2_, H1_, k0 + 32, (kk & 1) ? Ws0 : Ws1, tid);
      if (kk+1 < H1_/32) wait_async_le4(); else wait_async_0();
    } else {
      #pragma unroll
      for (int it = 0; it < 8; ++it) {
        int idx = (tid + it*256) * 4;
        int n = idx & 255, k = idx >> 8;
        float4 f = *(const float4*)(W2g + ((size_t)e*H1_ + k0 + k)*H2_ + n);
        Ws0[(n+0)*SW_ + k] = (_Float16)f.x;
        Ws0[(n+1)*SW_ + k] = (_Float16)f.y;
        Ws0[(n+2)*SW_ + k] = (_Float16)f.z;
        Ws0[(n+3)*SW_ + k] = (_Float16)f.w;
      }
    }
    __syncthreads();

    v16h a[4];
    #pragma unroll
    for (int i = 0; i < 4; ++i) {
      const _Float16* p = h1s + (wrow*64 + i*16 + mlane)*SH_ + k0 + hhi*8;
      a[i] = cat8(*(const v8h*)p, *(const v8h*)(p + 16));
    }
    #pragma unroll
    for (int j = 0; j < 4; ++j) {
      const _Float16* q = wsbuf + (wcol*64 + j*16 + mlane)*SW_ + hhi*8;
      v16h b = cat8(*(const v8h*)q, *(const v8h*)(q + 16));
      #pragma unroll
      for (int i = 0; i < 4; ++i)
        c[i][j] = __builtin_amdgcn_wmma_f32_16x16x32_f16(
            false, a[i], false, b, (short)0, c[i][j], false, false);
    }
  }

  // ---------------- layer 3: ssp(h2+b2) . W3 -> per-atom scalar --------
  float w3v[4], b2v[4];
  #pragma unroll
  for (int j = 0; j < 4; ++j) {
    int n = wcol*64 + j*16 + mlane;
    w3v[j] = W3g[e*H2_ + n];
    b2v[j] = b2g[e*H2_ + n];
  }
  #pragma unroll
  for (int i = 0; i < 4; ++i) {
    #pragma unroll
    for (int v = 0; v < 8; ++v) {
      float s = 0.0f;
      #pragma unroll
      for (int j = 0; j < 4; ++j)
        s += ssp(c[i][j][v] + b2v[j]) * w3v[j];
      // reduce the 16 lanes (same output row) within each half-wave
      #pragma unroll
      for (int off = 1; off < 16; off <<= 1)
        s += __shfl_xor(s, off, 16);
      if (mlane == 0)
        atomicAdd(&row_e[wrow*64 + i*16 + hhi*8 + v], s);
    }
  }
  __syncthreads();

  // one-hot select by atomic number, reduce tile -> batch energy
  if (tid < TM_) {
    int grow = tileRow + tid;
    float val = (Zg[grow] == Eg[e]) ? (row_e[tid] + b3g[e]) : 0.0f;
    atomicAdd(&bsum, val);    // STD=1, MEAN=0 -> identity affine
  }
  __syncthreads();
  if (tid == 0)
    atomicAdd(&outg[tileRow / N_], bsum);
}

extern "C" void kernel_launch(void* const* d_in, const int* in_sizes, int n_in,
                              void* d_out, int out_size, void* d_ws, size_t ws_size,
                              hipStream_t stream) {
  (void)in_sizes; (void)n_in;
  const float* rep   = (const float*)d_in[0];
  const int*   zn    = (const int*)d_in[1];
  const int*   elems = (const int*)d_in[2];
  const float* W1    = (const float*)d_in[3];
  const float* b1    = (const float*)d_in[4];
  const float* W2    = (const float*)d_in[5];
  const float* b2    = (const float*)d_in[6];
  const float* W3    = (const float*)d_in[7];
  const float* b3    = (const float*)d_in[8];
  float* out = (float*)d_out;

  zero_f32<<<1, 64, 0, stream>>>(out, out_size);

  const size_t need = (size_t)(WT1_ELEMS + WT2_ELEMS) * sizeof(_Float16);
  const bool useWt = (d_ws != nullptr) && (ws_size >= need);
  _Float16* W1t = (_Float16*)d_ws;
  _Float16* W2t = W1t + WT1_ELEMS;

  dim3 grid(BN_/TM_, E_);
  if (useWt) {
    int tot = WT1_ELEMS + WT2_ELEMS;
    prep_wt<<<(tot + 255)/256, 256, 0, stream>>>(W1, W2, W1t, W2t);
    (void)hipFuncSetAttribute((const void*)ani_mlp<true>,
                              hipFuncAttributeMaxDynamicSharedMemorySize,
                              (int)SMEM_BYTES);
    ani_mlp<true><<<grid, 256, SMEM_BYTES, stream>>>(
        rep, zn, elems, W1, b1, W2, b2, W3, b3, W1t, W2t, out);
  } else {
    (void)hipFuncSetAttribute((const void*)ani_mlp<false>,
                              hipFuncAttributeMaxDynamicSharedMemorySize,
                              (int)SMEM_BYTES);
    ani_mlp<false><<<grid, 256, SMEM_BYTES, stream>>>(
        rep, zn, elems, W1, b1, W2, b2, W3, b3, W1t, W2t, out);
  }
}
